// DowngradeProtocol_1898375545548
// MI455X (gfx1250) — matrix operand order repaired and checked by reading, not compiled, not run on hardware
//
#include <hip/hip_runtime.h>

typedef __attribute__((ext_vector_type(16))) _Float16 v16h;
typedef __attribute__((ext_vector_type(8)))  _Float16 v8h;
typedef __attribute__((ext_vector_type(8)))  float    v8f;

#define NB   4
#define NC   8
#define HH   1024
#define WW   1024
#define OUTD 256

#define XR      80    // x-region is 80x80 (5x5 tiles of 16)
#define IN_ROWS 88    // 80 + 8 (vertical taps of the 9x9)
#define IN_COLS 96    // 80 + 16 (last A-tile reads K=0..31 from col 64)
#define IN_STR  104   // halves; 208B row stride, 16B-aligned vector slots
#define XSTR    84    // floats

// _DEC12 taps of the second-stage filter (offsets -6..+5), symmetric
__device__ __constant__ float KK12[12] = {
    -1.20162964e-04f,  0.001615524292f, -0.010385513306f,  0.043619155884f,
    -0.145397186478f,  0.61066818237f,   0.61066818237f,  -0.145397186478f,
     0.043619155884f, -0.010385513306f,  0.001615524292f, -1.20162964e-04f
};

__global__ __launch_bounds__(256)
void mtf_resample_fused(const float* __restrict__ in,
                        const float* __restrict__ mtf,
                        const int*   __restrict__ rarr,
                        const int*   __restrict__ carr,
                        float*       __restrict__ out)
{
    __shared__ __align__(16) _Float16 sIn[IN_ROWS * IN_STR]; // staged input (f16)
    __shared__ __align__(16) _Float16 sB[9 * 16 * 32];       // Toeplitz B: [a][n][k]
    __shared__ float                  sX[XR * XSTR];         // x region (f32)

    const int tid = threadIdx.x;
    const int bid = blockIdx.x;
    const int tX  = bid & 15;
    const int tY  = (bid >> 4) & 15;
    const int ch  = (bid >> 8) & 7;
    const int bi  = bid >> 11;

    const int rv = rarr[bi * NC + ch];
    const int cv = carr[bi * NC + ch];
    const int ri = rv >> 1, rf = rv & 1;
    const int ci = cv >> 1, cf = cv & 1;

    // global coordinate of x-region (row0,col0); sampled center is +6 inside
    const int oy  = 2 + 64 * tY - ri - 6;
    const int ox  = 2 + 64 * tX - ci - 6;
    const int gy0 = oy - 4;   // staged-input origin (9x9 needs -4)
    const int gx0 = ox - 4;

    // ---- Stage 0: banded Toeplitz weights  sB[a][n][k] = mtf[a][k-n][ch], 0<=k-n<=8
    for (int idx = tid; idx < 9 * 16 * 32; idx += 256) {
        const int a = idx >> 9;
        const int n = (idx >> 5) & 15;
        const int k = idx & 31;
        const int d = k - n;
        float w = (d >= 0 && d <= 8) ? mtf[(a * 9 + d) * NC + ch] : 0.0f;
        sB[idx] = (_Float16)w;
    }

    // ---- Stage 1: stage input halo, f32 -> f16, clamped indices == edge padding
    const float* img = in + (size_t)(bi * NC + ch) * (size_t)(HH * WW);
    for (int idx = tid; idx < IN_ROWS * IN_COLS; idx += 256) {
        const int r0 = idx / IN_COLS, c0 = idx - r0 * IN_COLS;
        int gr = gy0 + r0; gr = gr < 0 ? 0 : (gr > HH - 1 ? HH - 1 : gr);
        int gc = gx0 + c0; gc = gc < 0 ? 0 : (gc > WW - 1 ? WW - 1 : gc);
        sIn[r0 * IN_STR + c0] = (_Float16)img[gr * WW + gc];
    }
    __syncthreads();

    // ---- Stage 2: 25 x-tiles (16x16) via v_wmma_f32_16x16x32_f16
    const int lane = tid & 31;
    const int wv   = tid >> 5;        // wave id 0..7
    const int m    = lane & 15;       // A-row / B-col / D-col
    const int hi16 = lane >> 4;       // lane group
    const int kAo  = hi16 * 8;        // A K-offsets: 0..7/16..23 vs 8..15/24..31
    const int kBo  = hi16 * 16;       // unused; B loads use kAo (same k order as A)

    for (int wt = wv; wt < 25; wt += 8) {
        const int ty = wt / 5, tx = wt - ty * 5;
        v8f acc = {};
        #pragma unroll
        for (int a = 0; a < 9; ++a) {
            // A[m,k] = staged_in[ty*16 + m + a][tx*16 + k]  (16x32 f16 layout)
            const _Float16* ap = &sIn[(ty * 16 + m + a) * IN_STR + tx * 16 + kAo];
            v8h alo = *(const v8h*)ap;          // K kAo..kAo+7
            v8h ahi = *(const v8h*)(ap + 16);   // K kAo+16..kAo+23
            v16h A = __builtin_shufflevector(alo, ahi,
                      0,1,2,3,4,5,6,7,8,9,10,11,12,13,14,15);
            const _Float16* bp = &sB[(a * 16 + m) * 32 + kAo];
            v8h blo = *(const v8h*)bp;          // K kAo..kAo+7
            v8h bhi = *(const v8h*)(bp + 16);   // K kAo+16..kAo+23
            v16h Bm = __builtin_shufflevector(blo, bhi,
                      0,1,2,3,4,5,6,7,8,9,10,11,12,13,14,15);
            acc = __builtin_amdgcn_wmma_f32_16x16x32_f16(
                      false, A, false, Bm, (short)0, acc, false, false);
        }
        // D layout: VGPR v -> row v (lanes 0-15) / v+8 (lanes 16-31), col = m
        #pragma unroll
        for (int v = 0; v < 8; ++v) {
            const int row = ty * 16 + v + hi16 * 8;
            sX[row * XSTR + tx * 16 + m] = acc[v];
        }
    }
    __syncthreads();

    // ---- Stage 3: parity-selected second filter at sampled points (f32)
    const int i = tid >> 4, j = tid & 15;
    const int by = 4 * i, bx = 4 * j;   // x-region index of (tap u=0, v=0); center +6
    float res;
    if (rf && cf) {                     // separable 12x12 (K_SE)
        res = 0.0f;
        for (int u = 0; u < 12; ++u) {
            const int gy = oy + by + u;
            if ((unsigned)gy < (unsigned)HH) {
                float s = 0.0f;
                #pragma unroll
                for (int v = 0; v < 12; ++v) {
                    const int gx = ox + bx + v;
                    if ((unsigned)gx < (unsigned)WW)
                        s += KK12[v] * sX[(by + u) * XSTR + bx + v];
                }
                res += KK12[u] * s;
            }
        }
    } else if (rf) {                    // vertical 12-tap (K_S)
        res = 0.0f;
        #pragma unroll
        for (int u = 0; u < 12; ++u) {
            const int gy = oy + by + u;
            if ((unsigned)gy < (unsigned)HH)
                res += KK12[u] * sX[(by + u) * XSTR + bx + 6];
        }
    } else if (cf) {                    // horizontal 12-tap (K_E)
        res = 0.0f;
        #pragma unroll
        for (int v = 0; v < 12; ++v) {
            const int gx = ox + bx + v;
            if ((unsigned)gx < (unsigned)WW)
                res += KK12[v] * sX[(by + 6) * XSTR + bx + v];
        }
    } else {                            // identity: x at sampled point
        res = sX[(by + 6) * XSTR + bx + 6];
    }

    out[((size_t)(bi * NC + ch) * OUTD + (tY * 16 + i)) * OUTD + (tX * 16 + j)] = res;
}

extern "C" void kernel_launch(void* const* d_in, const int* in_sizes, int n_in,
                              void* d_out, int out_size, void* d_ws, size_t ws_size,
                              hipStream_t stream) {
    (void)in_sizes; (void)n_in; (void)out_size; (void)d_ws; (void)ws_size;
    const float* outputs = (const float*)d_in[0];
    const float* mtf     = (const float*)d_in[1];
    const int*   r       = (const int*)d_in[2];
    const int*   c       = (const int*)d_in[3];
    float*       out     = (float*)d_out;

    const int nblocks = NB * NC * 16 * 16;   // 8192 (b, ch, tileY, tileX)
    mtf_resample_fused<<<nblocks, 256, 0, stream>>>(outputs, mtf, r, c, out);
}
